// VRNN_46969762349590
// MI455X (gfx1250) — compile-verified, matrix operand order checked by reading, not run on hardware
//
#include <hip/hip_runtime.h>
#include <hip/hip_bf16.h>
#include <cstdint>

// ---------------------------------------------------------------------------
// VRNN on gfx1250 (MI455X): wave32, WMMA f32<=f16 16x16x32 everywhere.
// ---------------------------------------------------------------------------

#define SEQ   400
#define BATCH 256
#define XDIM  513
#define XPAD  528      // Wdv output-dim padded to multiple of 16
#define KXPAD 544      // x / Wx K-dim padded to multiple of 32
#define ZDIM  16
#define HID   128
#define LDA   40       // padded LDS row stride (halves) for fx staging tiles

// sched_group_barrier masks
#define SGB_WMMA  0x008
#define SGB_VMRD  0x020
#define SGB_DSRD  0x100
#define SGB_DSWR  0x200

typedef _Float16 v16h __attribute__((ext_vector_type(16)));
typedef _Float16 v8h  __attribute__((ext_vector_type(8)));
typedef float    v8f  __attribute__((ext_vector_type(8)));

// ---- WMMA helpers ---------------------------------------------------------
// A fragment (16 rows x K=32 f16) and B fragment (W.T, i.e. W row-major
// [N][K], lane = output column) share the same per-lane addressing per ISA
// 7.12.2: lanes 0-15 -> K {k0..k0+7, k0+16..k0+23}; lanes 16-31 -> +8.
__device__ __forceinline__ v16h load_frag(const _Float16* base, int ld, int k0) {
    const int lane = threadIdx.x & 31;
    const int r    = lane & 15;
    const int kb   = k0 + ((lane & 16) ? 8 : 0);
    const _Float16* p = base + (size_t)r * ld + kb;
    v8h lo = *(const v8h*)(p);
    v8h hi = *(const v8h*)(p + 16);
    return __builtin_shufflevector(lo, hi, 0, 1, 2, 3, 4, 5, 6, 7,
                                           8, 9, 10, 11, 12, 13, 14, 15);
}

__device__ __forceinline__ v8f wmma16(v16h a, v16h b, v8f c) {
    // D = A x B + C, f32 accumulate. (neg_a, A, neg_b, B, c_mod, C, reuseA, reuseB)
    return __builtin_amdgcn_wmma_f32_16x16x32_f16(false, a, false, b,
                                                  (short)0, c, false, false);
}

// GEMM over NCH K-chunks of 32 with preloaded A fragments. All B fragments
// are preloaded, and a sched_group_barrier pipeline [2*NCH VMEM reads] ->
// [NCH WMMAs] keeps the loads in a clause with distinct registers so the
// waits retire incrementally (verified in round-4 disassembly).
template <int NCH>
__device__ __forceinline__ v8f gemm_acc(v8f acc, const v16h* af,
                                        const _Float16* W, int ld) {
    v16h bf[NCH];
#pragma unroll
    for (int j = 0; j < NCH; ++j) bf[j] = load_frag(W, ld, j * 32);
#pragma unroll
    for (int j = 0; j < NCH; ++j) acc = wmma16(af[j], bf[j], acc);
    __builtin_amdgcn_sched_group_barrier(SGB_VMRD, 2 * NCH, 0);
    __builtin_amdgcn_sched_group_barrier(SGB_WMMA, NCH, 0);
    return acc;
}

// C/D layout: VGPR i -> row (i + (lane<16 ? 0 : 8)), col = lane & 15.
__device__ __forceinline__ void cstore_f16(_Float16* base, int ld, v8f c) {
    const int lane = threadIdx.x & 31;
    const int n  = lane & 15;
    const int mo = (lane & 16) ? 8 : 0;
#pragma unroll
    for (int i = 0; i < 8; ++i) base[(size_t)(i + mo) * ld + n] = (_Float16)c[i];
}

__device__ __forceinline__ float sigmoidf_(float x) {
    return 1.0f / (1.0f + __expf(-x));
}

// ---------------------------------------------------------------------------
// K0: convert f32 weight [nsrc][ksrc] -> f16 [npad][kpad], zero padded.
// ---------------------------------------------------------------------------
__global__ void wconv_kernel(const float* __restrict__ src, _Float16* __restrict__ dst,
                             int nsrc, int npad, int ksrc, int kpad) {
    size_t total = (size_t)npad * kpad;
    for (size_t i = (size_t)blockIdx.x * blockDim.x + threadIdx.x; i < total;
         i += (size_t)gridDim.x * blockDim.x) {
        int nn = (int)(i / kpad);
        int kk = (int)(i % kpad);
        float v = (nn < nsrc && kk < ksrc) ? src[(size_t)nn * ksrc + kk] : 0.0f;
        dst[i] = (_Float16)v;
    }
}

// ---------------------------------------------------------------------------
// K1: fx[s][b][h] = relu( x[b][:][s] @ Wx.T + bx )   (f16 result)
// Block = 16b x 16s row panel (256 GEMM rows), K looped in 32-chunks with
// double-buffered LDS staging. Staging is branchless (clamped address +
// select) and split into a fully unrolled load-to-registers phase followed
// by convert+store, so the 16 x-loads per thread batch into a clause instead
// of a per-element load->wait(0) latency chain. Lanes 0-15 read 16
// consecutive s (64B coalesced segments).
// grid = (SEQ/16)*(BATCH/16) = 400 blocks, 512 threads (16 waves).
// ---------------------------------------------------------------------------
__global__ __launch_bounds__(512) void fx_kernel(const float* __restrict__ x,
                                                 const _Float16* __restrict__ Wx,   // [HID][KXPAD]
                                                 const float* __restrict__ bx,
                                                 _Float16* __restrict__ fx) {
    __shared__ __align__(16) _Float16 sA[2][256 * LDA];
    const int s0 = (blockIdx.x >> 4) << 4;   // 0..384
    const int b0 = (blockIdx.x & 15) << 4;   // 0..240
    const int wave = threadIdx.x >> 5;       // 0..15 -> b-row-tile
    const int lane = threadIdx.x & 31;
    const int n    = lane & 15;
    const int mo   = (lane & 16) ? 8 : 0;

    // stage chunk: sA[buf][row*LDA + kk] = x[b0+row/16][k0+kk][s0+row%16]
    // branchless tail handling: clamp k, select 0 beyond XDIM.
    auto stage = [&](int chunk, _Float16* dst) {
        const int k0 = chunk * 32;
        float v[16];
#pragma unroll
        for (int j = 0; j < 16; ++j) {
            const int idx = (int)threadIdx.x + j * 512;
            const int row = idx & 255;   // consecutive lanes -> consecutive s
            const int kk  = k0 + (idx >> 8);
            const int kc  = kk < XDIM ? kk : (XDIM - 1);
            float t = x[((size_t)(b0 + (row >> 4)) * XDIM + kc) * SEQ + s0 + (row & 15)];
            v[j] = (kk < XDIM) ? t : 0.0f;
        }
#pragma unroll
        for (int j = 0; j < 16; ++j) {
            const int idx = (int)threadIdx.x + j * 512;
            dst[(idx & 255) * LDA + (idx >> 8)] = (_Float16)v[j];
        }
    };

    stage(0, sA[0]);
    __syncthreads();

    v8f acc[8];
#pragma unroll
    for (int nt = 0; nt < 8; ++nt) acc[nt] = (v8f){};

    const int nchunks = KXPAD / 32;   // 17
    for (int c = 0; c < nchunks; ++c) {
        const _Float16* buf = sA[c & 1];
        v16h a = load_frag(buf + (size_t)wave * 16 * LDA, LDA, 0);
        v16h bf[8];
#pragma unroll
        for (int nt = 0; nt < 8; ++nt)
            bf[nt] = load_frag(Wx + (size_t)(nt * 16) * KXPAD, KXPAD, c * 32);
#pragma unroll
        for (int nt = 0; nt < 8; ++nt)
            acc[nt] = wmma16(a, bf[nt], acc[nt]);
        if (c + 1 < nchunks) {
            stage(c + 1, sA[(c + 1) & 1]);
            // pipeline: [A ds reads] -> [16 B + 16 x global reads] ->
            //           [8 WMMAs] -> [16 stage ds writes]
            __builtin_amdgcn_sched_group_barrier(SGB_DSRD, 2, 0);
            __builtin_amdgcn_sched_group_barrier(SGB_VMRD, 32, 0);
            __builtin_amdgcn_sched_group_barrier(SGB_WMMA, 8, 0);
            __builtin_amdgcn_sched_group_barrier(SGB_DSWR, 16, 0);
        } else {
            __builtin_amdgcn_sched_group_barrier(SGB_DSRD, 2, 0);
            __builtin_amdgcn_sched_group_barrier(SGB_VMRD, 16, 0);
            __builtin_amdgcn_sched_group_barrier(SGB_WMMA, 8, 0);
        }
        __syncthreads();
    }

    const int b = b0 + wave;
#pragma unroll
    for (int nt = 0; nt < 8; ++nt) {
        const float bv = bx[nt * 16 + n];
#pragma unroll
        for (int i = 0; i < 8; ++i) {
            float v = acc[nt][i] + bv;
            v = v > 0.0f ? v : 0.0f;
            fx[((size_t)(s0 + i + mo) * BATCH + b) * HID + nt * 16 + n] = (_Float16)v;
        }
    }
}

// ---------------------------------------------------------------------------
// K2: sequential scan. 16 blocks (one per 16-row batch slice), 8 waves each.
// h carried as f16 (matmul input), c carried as f32 (additive path).
// ---------------------------------------------------------------------------
__global__ __launch_bounds__(256) void scan_kernel(
    const _Float16* __restrict__ fx,      // [S][B][HID] f16
    const float*    __restrict__ eps,     // [S][B][Z]   f32
    const _Float16* __restrict__ Wenc, const float* __restrict__ benc,  // [HID][256]
    const _Float16* __restrict__ Wem,  const float* __restrict__ bem,   // [Z][HID]
    const _Float16* __restrict__ Wev,  const float* __restrict__ bev,   // [Z][HID]
    const _Float16* __restrict__ Wz,   const float* __restrict__ bz,    // [HID][32]
    const _Float16* __restrict__ Wdec, const float* __restrict__ bdec,  // [HID][256]
    const _Float16* __restrict__ Wdv,  const float* __restrict__ bdv,   // [XPAD][HID]
    const _Float16* __restrict__ Wih,  const float* __restrict__ bih,   // [512][256]
    const _Float16* __restrict__ Whh,  const float* __restrict__ bhh,   // [512][HID]
    _Float16* __restrict__ hseq,          // [S][B][HID] f16 (pre-update h)
    float* __restrict__ y_out,            // [B][X][S]
    float* __restrict__ mean_out,         // [S][B][Z]
    float* __restrict__ logvar_out,       // [S][B][Z]
    float* __restrict__ z_out)            // [B][Z][S]
{
    __shared__ __align__(16) _Float16 sh_h[16 * HID];
    __shared__ __align__(16) float    sh_c[16 * HID];
    __shared__ __align__(16) _Float16 sh_e[16 * HID];
    __shared__ __align__(16) _Float16 sh_fz[16 * HID];
    __shared__ __align__(16) _Float16 sh_d[16 * HID];
    __shared__ __align__(16) _Float16 sh_z[16 * 32];
    __shared__ __align__(16) float    sh_mn[16 * ZDIM];
    __shared__ __align__(16) float    sh_lv[16 * ZDIM];

    const int b0   = blockIdx.x << 4;
    const int wave = threadIdx.x >> 5;
    const int lane = threadIdx.x & 31;
    const int n    = lane & 15;
    const int mo   = (lane & 16) ? 8 : 0;

    for (int i = threadIdx.x; i < 16 * HID; i += 256) {
        sh_h[i] = (_Float16)0.0f;
        sh_c[i] = 0.0f;
    }
    for (int i = threadIdx.x; i < 16 * 32; i += 256) sh_z[i] = (_Float16)0.0f;
    __syncthreads();

    for (int t = 0; t < SEQ; ++t) {
        const _Float16* fxt = fx + ((size_t)t * BATCH + b0) * HID;

        // prefetch next timestep's fx tile (global_prefetch_b8 path)
        if (t + 1 < SEQ)
            __builtin_prefetch(fx + ((size_t)(t + 1) * BATCH + b0) * HID
                                  + (size_t)threadIdx.x * 8, 0, 3);

        // hseq[t] = h (pre-update)
        for (int i = threadIdx.x; i < 16 * HID; i += 256)
            hseq[((size_t)t * BATCH + b0) * HID + i] = sh_h[i];

        // ---- enc: e = relu([fxt, h] @ Wenc.T + benc) ----
        {
            const int n0 = wave * 16;
            v16h af[8];
#pragma unroll
            for (int k = 0; k < 4; ++k) af[k] = load_frag(fxt, HID, k * 32);
#pragma unroll
            for (int k = 0; k < 4; ++k) af[4 + k] = load_frag(sh_h, HID, k * 32);
            v8f acc = {};
            acc = gemm_acc<8>(acc, af, Wenc + (size_t)n0 * 256, 256);
            const float bv = benc[n0 + n];
#pragma unroll
            for (int i = 0; i < 8; ++i) {
                float v = acc[i] + bv;
                acc[i] = v > 0.0f ? v : 0.0f;
            }
            cstore_f16(sh_e + n0, HID, acc);
        }
        __syncthreads();

        // ---- mean / logvar heads (waves 0 and 1) ----
        if (wave < 2) {
            const _Float16* W  = wave ? Wev : Wem;
            const float*    bb = wave ? bev : bem;
            float* gout = wave ? logvar_out : mean_out;
            float* shd  = wave ? sh_lv : sh_mn;
            v16h af[4];
#pragma unroll
            for (int k = 0; k < 4; ++k) af[k] = load_frag(sh_e, HID, k * 32);
            v8f acc = {};
            acc = gemm_acc<4>(acc, af, W, HID);
            const float bv = bb[n];
#pragma unroll
            for (int i = 0; i < 8; ++i) {
                float v = acc[i] + bv;
                shd[(i + mo) * ZDIM + n] = v;
                gout[((size_t)t * BATCH + b0 + i + mo) * ZDIM + n] = v;
            }
        }
        __syncthreads();

        // ---- reparameterize: z = eps * exp(0.5*logvar) + mean ----
        {
            const int r = threadIdx.x >> 4;   // 0..15 batch row
            const int j = threadIdx.x & 15;   // 0..15 z dim
            float lv = sh_lv[r * ZDIM + j];
            float mn = sh_mn[r * ZDIM + j];
            float zz = eps[((size_t)t * BATCH + b0 + r) * ZDIM + j] * __expf(0.5f * lv) + mn;
            z_out[((size_t)(b0 + r) * ZDIM + j) * SEQ + t] = zz;
            sh_z[r * 32 + j] = (_Float16)zz;  // cols 16..31 stay zero
        }
        __syncthreads();

        // ---- fz = relu(z @ Wz.T + bz) : single K=32 WMMA per wave ----
        {
            const int n0 = wave * 16;
            v8f acc = {};
            acc = wmma16(load_frag(sh_z, 32, 0),
                         load_frag(Wz + (size_t)n0 * 32, 32, 0), acc);
            const float bv = bz[n0 + n];
#pragma unroll
            for (int i = 0; i < 8; ++i) {
                float v = acc[i] + bv;
                acc[i] = v > 0.0f ? v : 0.0f;
            }
            cstore_f16(sh_fz + n0, HID, acc);
        }
        __syncthreads();

        // ---- dec: d = relu([fz, h] @ Wdec.T + bdec) ----
        {
            const int n0 = wave * 16;
            v16h af[8];
#pragma unroll
            for (int k = 0; k < 4; ++k) af[k] = load_frag(sh_fz, HID, k * 32);
#pragma unroll
            for (int k = 0; k < 4; ++k) af[4 + k] = load_frag(sh_h, HID, k * 32);
            v8f acc = {};
            acc = gemm_acc<8>(acc, af, Wdec + (size_t)n0 * 256, 256);
            const float bv = bdec[n0 + n];
#pragma unroll
            for (int i = 0; i < 8; ++i) {
                float v = acc[i] + bv;
                acc[i] = v > 0.0f ? v : 0.0f;
            }
            cstore_f16(sh_d + n0, HID, acc);
        }
        __syncthreads();

        // ---- y = exp(d @ Wdv.T + bdv), stored transposed [B][X][S] ----
        {
            v16h df[4];
#pragma unroll
            for (int k = 0; k < 4; ++k) df[k] = load_frag(sh_d, HID, k * 32);
            for (int nt = wave; nt < 33; nt += 8) {   // 33 tiles of 16 cols (528 padded)
                const int n0 = nt * 16;
                v8f acc = {};
                acc = gemm_acc<4>(acc, df, Wdv + (size_t)n0 * HID, HID);
                const int col = n0 + n;
                if (col < XDIM) {
                    const float bv = bdv[col];
#pragma unroll
                    for (int i = 0; i < 8; ++i)
                        y_out[((size_t)(b0 + i + mo) * XDIM + col) * SEQ + t] =
                            __expf(acc[i] + bv);
                }
            }
        }

        // ---- gates = [fxt, fz] @ Wih.T + bih + h @ Whh.T + bhh ----
        // A fragments hoisted out of the gate loop: af[0..3]=fxt, af[4..7]=fz
        // (K of Wih), af[8..11]=h (K of Whh). wave w owns hidden cols
        // [16w,16w+16) in all four gate regions: tile w->i, w+8->f, w+16->g,
        // w+24->o, so the LSTM update stays in registers.
        v8f g[4];
        {
            v16h af[12];
#pragma unroll
            for (int k = 0; k < 4; ++k) af[k]     = load_frag(fxt, HID, k * 32);
#pragma unroll
            for (int k = 0; k < 4; ++k) af[4 + k] = load_frag(sh_fz, HID, k * 32);
#pragma unroll
            for (int k = 0; k < 4; ++k) af[8 + k] = load_frag(sh_h, HID, k * 32);
#pragma unroll
            for (int gi = 0; gi < 4; ++gi) {
                const int n0 = (wave + gi * 8) * 16;
                v8f acc = {};
                acc = gemm_acc<8>(acc, af,     Wih + (size_t)n0 * 256, 256);
                acc = gemm_acc<4>(acc, af + 8, Whh + (size_t)n0 * HID, HID);
                const float bv = bih[n0 + n] + bhh[n0 + n];
#pragma unroll
                for (int i = 0; i < 8; ++i) acc[i] += bv;
                g[gi] = acc;
            }
        }
        __syncthreads();   // everyone done reading old h/fz/d

        // ---- LSTM cell update (pure registers + LDS state) ----
        {
            const int j = wave * 16 + n;
#pragma unroll
            for (int i = 0; i < 8; ++i) {
                const int idx = (i + mo) * HID + j;
                float ig = sigmoidf_(g[0][i]);
                float fg = sigmoidf_(g[1][i]);
                float gg = tanhf(g[2][i]);
                float og = sigmoidf_(g[3][i]);
                float cn = fg * sh_c[idx] + ig * gg;
                sh_c[idx] = cn;
                sh_h[idx] = (_Float16)(og * tanhf(cn));
            }
        }
        __syncthreads();
    }
}

// ---------------------------------------------------------------------------
// K3: prior path: p = relu(hseq @ Wpr.T + bpr); prior heads via Wem/Wev.
// grid = SEQ*16 blocks, 8 waves.
// ---------------------------------------------------------------------------
__global__ __launch_bounds__(256) void prior_kernel(
    const _Float16* __restrict__ hseq,    // [S][B][HID] f16
    const _Float16* __restrict__ Wpr, const float* __restrict__ bpr,  // [HID][HID]
    const _Float16* __restrict__ Wem, const float* __restrict__ bem,
    const _Float16* __restrict__ Wev, const float* __restrict__ bev,
    float* __restrict__ meanp_out,        // [S][B][Z]
    float* __restrict__ logvarp_out)      // [S][B][Z]
{
    __shared__ __align__(16) _Float16 sP[16 * HID];
    const int t    = blockIdx.x >> 4;
    const int b0   = (blockIdx.x & 15) << 4;
    const int wave = threadIdx.x >> 5;
    const int lane = threadIdx.x & 31;
    const int n    = lane & 15;
    const int mo   = (lane & 16) ? 8 : 0;

    const _Float16* A = hseq + ((size_t)t * BATCH + b0) * HID;
    {
        const int n0 = wave * 16;
        v16h af[4];
#pragma unroll
        for (int k = 0; k < 4; ++k) af[k] = load_frag(A, HID, k * 32);
        v8f acc = {};
        acc = gemm_acc<4>(acc, af, Wpr + (size_t)n0 * HID, HID);
        const float bv = bpr[n0 + n];
#pragma unroll
        for (int i = 0; i < 8; ++i) {
            float v = acc[i] + bv;
            acc[i] = v > 0.0f ? v : 0.0f;
        }
        cstore_f16(sP + n0, HID, acc);
    }
    __syncthreads();

    if (wave < 2) {
        const _Float16* W  = wave ? Wev : Wem;
        const float*    bb = wave ? bev : bem;
        float* out = wave ? logvarp_out : meanp_out;
        v16h af[4];
#pragma unroll
        for (int k = 0; k < 4; ++k) af[k] = load_frag(sP, HID, k * 32);
        v8f acc = {};
        acc = gemm_acc<4>(acc, af, W, HID);
        const float bv = bb[n];
#pragma unroll
        for (int i = 0; i < 8; ++i)
            out[((size_t)t * BATCH + b0 + i + mo) * ZDIM + n] = acc[i] + bv;
    }
}

// ---------------------------------------------------------------------------
// Launch
// ---------------------------------------------------------------------------
extern "C" void kernel_launch(void* const* d_in, const int* in_sizes, int n_in,
                              void* d_out, int out_size, void* d_ws, size_t ws_size,
                              hipStream_t stream) {
    const float* x    = (const float*)d_in[0];
    const float* eps  = (const float*)d_in[1];
    const float* Wx   = (const float*)d_in[2];   const float* bx   = (const float*)d_in[3];
    const float* Wzf  = (const float*)d_in[4];   const float* bz   = (const float*)d_in[5];
    const float* Wenc = (const float*)d_in[6];   const float* benc = (const float*)d_in[7];
    const float* Wem  = (const float*)d_in[8];   const float* bem  = (const float*)d_in[9];
    const float* Wev  = (const float*)d_in[10];  const float* bev  = (const float*)d_in[11];
    const float* Wdec = (const float*)d_in[12];  const float* bdec = (const float*)d_in[13];
    const float* Wdv  = (const float*)d_in[14];  const float* bdv  = (const float*)d_in[15];
    const float* Wpr  = (const float*)d_in[16];  const float* bpr  = (const float*)d_in[17];
    const float* Wih  = (const float*)d_in[18];  const float* bih  = (const float*)d_in[19];
    const float* Whh  = (const float*)d_in[20];  const float* bhh  = (const float*)d_in[21];

    // ---- workspace carve (256B aligned) ----
    char* wsp = (char*)d_ws;
    auto carve = [&](size_t bytes) -> void* {
        void* p = (void*)wsp;
        wsp += (bytes + 255) & ~(size_t)255;
        return p;
    };
    _Float16* fx16   = (_Float16*)carve((size_t)SEQ * BATCH * HID * 2);
    _Float16* hseq16 = (_Float16*)carve((size_t)SEQ * BATCH * HID * 2);
    _Float16* Wx16   = (_Float16*)carve((size_t)HID * KXPAD * 2);
    _Float16* Wenc16 = (_Float16*)carve((size_t)HID * 256 * 2);
    _Float16* Wem16  = (_Float16*)carve((size_t)ZDIM * HID * 2);
    _Float16* Wev16  = (_Float16*)carve((size_t)ZDIM * HID * 2);
    _Float16* Wz16   = (_Float16*)carve((size_t)HID * 32 * 2);
    _Float16* Wdec16 = (_Float16*)carve((size_t)HID * 256 * 2);
    _Float16* Wdv16  = (_Float16*)carve((size_t)XPAD * HID * 2);
    _Float16* Wih16  = (_Float16*)carve((size_t)4 * HID * 256 * 2);
    _Float16* Whh16  = (_Float16*)carve((size_t)4 * HID * HID * 2);
    _Float16* Wpr16  = (_Float16*)carve((size_t)HID * HID * 2);

    auto conv = [&](const float* src, _Float16* dst, int nsrc, int npad, int ksrc, int kpad) {
        size_t total = (size_t)npad * kpad;
        int blocks = (int)((total + 255) / 256);
        if (blocks > 1024) blocks = 1024;
        wconv_kernel<<<blocks, 256, 0, stream>>>(src, dst, nsrc, npad, ksrc, kpad);
    };
    conv(Wx,   Wx16,   HID,      HID,      XDIM, KXPAD);
    conv(Wenc, Wenc16, HID,      HID,      256,  256);
    conv(Wem,  Wem16,  ZDIM,     ZDIM,     HID,  HID);
    conv(Wev,  Wev16,  ZDIM,     ZDIM,     HID,  HID);
    conv(Wzf,  Wz16,   HID,      HID,      ZDIM, 32);
    conv(Wdec, Wdec16, HID,      HID,      256,  256);
    conv(Wdv,  Wdv16,  XDIM,     XPAD,     HID,  HID);
    conv(Wih,  Wih16,  4 * HID,  4 * HID,  256,  256);
    conv(Whh,  Whh16,  4 * HID,  4 * HID,  HID,  HID);
    conv(Wpr,  Wpr16,  HID,      HID,      HID,  HID);

    // ---- output carve (reference tuple order) ----
    float* out         = (float*)d_out;
    float* y_out       = out;                                            // [B][X][S]
    float* mean_out    = y_out + (size_t)BATCH * XDIM * SEQ;             // [S][B][Z]
    float* logvar_out  = mean_out + (size_t)SEQ * BATCH * ZDIM;
    float* meanp_out   = logvar_out + (size_t)SEQ * BATCH * ZDIM;
    float* logvarp_out = meanp_out + (size_t)SEQ * BATCH * ZDIM;
    float* z_out       = logvarp_out + (size_t)SEQ * BATCH * ZDIM;       // [B][Z][S]

    fx_kernel<<<(SEQ / 16) * (BATCH / 16), 512, 0, stream>>>(x, Wx16, bx, fx16);

    scan_kernel<<<BATCH / 16, 256, 0, stream>>>(
        fx16, eps,
        Wenc16, benc, Wem16, bem, Wev16, bev, Wz16, bz,
        Wdec16, bdec, Wdv16, bdv, Wih16, bih, Whh16, bhh,
        hseq16, y_out, mean_out, logvar_out, z_out);

    prior_kernel<<<SEQ * 16, 256, 0, stream>>>(
        hseq16, Wpr16, bpr, Wem16, bem, Wev16, bev, meanp_out, logvarp_out);
}
